// DenseFrameEmbedding_1906965479580
// MI455X (gfx1250) — matrix operand order repaired and checked by reading, not compiled, not run on hardware
//
#include <hip/hip_runtime.h>

#define B_ 8
#define CIN_ 2
#define BINS_ 1024
#define FRAMES_ 512
#define C_ 8
#define E_ 1024
#define N_ (B_ * FRAMES_)   // 4096 rows per channel

typedef __attribute__((ext_vector_type(16))) __bf16 v16bf;
typedef __attribute__((ext_vector_type(8)))  float  v8f;
typedef __attribute__((ext_vector_type(4)))  unsigned int u32x4;

__device__ __forceinline__ unsigned short f2bf(float f) {
    unsigned u = __builtin_bit_cast(unsigned, f);
    u += 0x7FFFu + ((u >> 16) & 1u);          // round-to-nearest-even
    return (unsigned short)(u >> 16);
}
__device__ __forceinline__ float bf2f(unsigned short h) {
    return __builtin_bit_cast(float, (unsigned)h << 16);
}

// ---------------------------------------------------------------------------
// Kernel P: pack codebooks f32 -> bf16, compute e2[c][e] = sum_k cb^2, zero loss
// ---------------------------------------------------------------------------
__global__ void pack_cb_kernel(const float* __restrict__ cb,
                               unsigned short* __restrict__ cbbf,
                               float* __restrict__ e2,
                               float* __restrict__ lossPtr) {
    int row = blockIdx.x;                 // c*E + e
    int tid = threadIdx.x;
    const float* src = cb + (size_t)row * BINS_;
    unsigned short* dst = cbbf + (size_t)row * BINS_;
    float s = 0.f;
#pragma unroll
    for (int i = 0; i < 4; ++i) {
        int k = tid + i * 256;
        float v = src[k];
        dst[k] = f2bf(v);
        s += v * v;
    }
    for (int off = 16; off > 0; off >>= 1) s += __shfl_xor(s, off, 32);
    __shared__ float red[8];
    if ((tid & 31) == 0) red[tid >> 5] = s;
    __syncthreads();
    if (tid == 0) {
        float t = 0.f;
#pragma unroll
        for (int i = 0; i < 8; ++i) t += red[i];
        e2[row] = t;
        if (row == 0) *lossPtr = 0.f;
    }
}

// ---------------------------------------------------------------------------
// Kernel A: conv1(9x1,pad4) + leaky_relu(0.2) + conv2(9x1,pad4) + skip(1x1)
// One block per (b,frame) column; writes v bf16 in [c][n][bin] layout.
// ---------------------------------------------------------------------------
__global__ void conv_kernel(const float* __restrict__ x,
                            const float* __restrict__ w1, const float* __restrict__ b1,
                            const float* __restrict__ w2, const float* __restrict__ b2,
                            const float* __restrict__ sw, const float* __restrict__ sb,
                            unsigned short* __restrict__ vbf) {
    int n = blockIdx.x;                   // 0..4095 = b*512 + frame
    int b = n >> 9, frame = n & 511;
    int tid = threadIdx.x;                // 256
    __shared__ float xs[CIN_][BINS_ + 8];
    __shared__ float hs[C_][BINS_ + 8];

    for (int ci = 0; ci < CIN_; ++ci) {
        for (int bin = tid; bin < BINS_; bin += 256)
            xs[ci][bin + 4] = x[(((size_t)b * CIN_ + ci) * BINS_ + bin) * FRAMES_ + frame];
        if (tid < 4) { xs[ci][tid] = 0.f; xs[ci][BINS_ + 4 + tid] = 0.f; }
    }
    __syncthreads();

    for (int c = 0; c < C_; ++c) {
        float wr[CIN_][9];
#pragma unroll
        for (int ci = 0; ci < CIN_; ++ci)
#pragma unroll
            for (int t = 0; t < 9; ++t) wr[ci][t] = w1[(c * CIN_ + ci) * 9 + t];
        float bb = b1[c];
        for (int bin = tid; bin < BINS_; bin += 256) {
            float a = bb;
#pragma unroll
            for (int ci = 0; ci < CIN_; ++ci)
#pragma unroll
                for (int t = 0; t < 9; ++t) a += wr[ci][t] * xs[ci][bin + t];
            hs[c][bin + 4] = (a > 0.f) ? a : 0.2f * a;   // leaky relu
        }
        if (tid < 4) { hs[c][tid] = 0.f; hs[c][BINS_ + 4 + tid] = 0.f; }
    }
    __syncthreads();

    for (int c = 0; c < C_; ++c) {
        float wr[C_][9];
#pragma unroll
        for (int cj = 0; cj < C_; ++cj)
#pragma unroll
            for (int t = 0; t < 9; ++t) wr[cj][t] = w2[(c * C_ + cj) * 9 + t];
        float s0 = sw[c * CIN_ + 0], s1 = sw[c * CIN_ + 1];
        float bb = b2[c] + sb[c];
        unsigned short* dst = vbf + ((size_t)c * N_ + n) * BINS_;
        for (int bin = tid; bin < BINS_; bin += 256) {
            float a = bb + s0 * xs[0][bin + 4] + s1 * xs[1][bin + 4];
#pragma unroll
            for (int cj = 0; cj < C_; ++cj)
#pragma unroll
                for (int t = 0; t < 9; ++t) a += wr[cj][t] * hs[cj][bin + t];
            dst[bin] = f2bf(a);
        }
    }
}

// ---------------------------------------------------------------------------
// Kernel B: per-channel GEMM (v · cb^T) via v_wmma_f32_16x16x32_bf16.
// Fully unrolled K loop (straight-line 512 WMMAs, all loads base+imm-offset),
// 8-deep register-pipelined B fragments, ping-pong A fragments.
// Fused argmin_e(e2 - 2*dot).
// Block = 8 waves; wave = 16M x 256N; block = 32M x 1024N (full E per block).
// ---------------------------------------------------------------------------
typedef union { u32x4 u[2]; v16bf v; } Frag;

__global__ void __launch_bounds__(256)
gemm_argmin_kernel(const unsigned short* __restrict__ vbf,
                   const unsigned short* __restrict__ cbbf,
                   const float* __restrict__ e2,
                   int* __restrict__ idxOut) {
    int c   = blockIdx.x;                 // channel
    int mb  = blockIdx.y;                 // 32-row block
    int tid = threadIdx.x;
    int wave = tid >> 5, lane = tid & 31;
    int mtile = wave >> 2, nq = wave & 3;
    int g = lane >> 4, nl = lane & 15;
    int mrow  = mb * 32 + mtile * 16 + nl;    // A row this lane loads
    int nbase = nq * 256;

    const u32x4* arow = (const u32x4*)(vbf  + ((size_t)c * N_ + mrow) * BINS_);
    const u32x4* brow = (const u32x4*)(cbbf + ((size_t)c * E_ + nbase + nl) * BINS_);

    v8f zero = {};
    v8f acc[16];
#pragma unroll
    for (int t = 0; t < 16; ++t) acc[t] = zero;

    // ---- prologue: A fragment for kk=0, B tiles 0..7 of kk=0 ----
    Frag aA, aB, bmb[8];
    aA.u[0] = arow[2 * g];
    aA.u[1] = arow[2 * g + 1];
    {
        int ib0 = 2 * g;
#pragma unroll
        for (int t = 0; t < 8; ++t) {
            bmb[t].u[0] = brow[ib0 + t * 2048];       // 16 rows * 128 u32x4/row
            bmb[t].u[1] = brow[ib0 + t * 2048 + 1];
        }
    }

    // one 32-wide K step: 16 WMMAs; refills keep 7 B fragments in flight.
    auto kstep = [&](int kk, Frag& aUse, Frag& aPre, bool hasNext) {
        int ibc = (kk >> 3) + 2 * g;      // this kk's B base index
        int ibn = ibc + 4;                // next kk's B base index
#pragma unroll
        for (int t = 0; t < 16; ++t) {
            acc[t] = __builtin_amdgcn_wmma_f32_16x16x32_bf16(
                false, aUse.v, false, bmb[t & 7].v, (short)0, acc[t], false, false);
            if (t == 0 && hasNext) {      // prefetch next kk's A fragment
                int ian = ((kk + 32) >> 3) + 2 * g;
                aPre.u[0] = arow[ian];
                aPre.u[1] = arow[ian + 1];
            }
            if (t < 8) {                  // refill slot with tile t+8 of this kk
                int tt = t + 8;
                bmb[t & 7].u[0] = brow[ibc + tt * 2048];
                bmb[t & 7].u[1] = brow[ibc + tt * 2048 + 1];
            } else if (hasNext) {         // refill with tile t-8 of next kk
                int tt = t - 8;
                bmb[t & 7].u[0] = brow[ibn + tt * 2048];
                bmb[t & 7].u[1] = brow[ibn + tt * 2048 + 1];
            }
        }
    };

#pragma unroll
    for (int kk = 0; kk < BINS_ - 64; kk += 64) {   // fully unrolled: straight line
        kstep(kk,      aA, aB, true);
        kstep(kk + 32, aB, aA, true);
    }
    kstep(BINS_ - 64, aA, aB, true);                // peeled tail
    kstep(BINS_ - 32, aB, aA, false);

    // ---- argmin over E: d = e2[n] - 2*dot ----
    __shared__ float cval[32][4];
    __shared__ int   cidx[32][4];
    float e2v[16];
#pragma unroll
    for (int t = 0; t < 16; ++t) e2v[t] = e2[c * E_ + nbase + t * 16 + nl];

#pragma unroll
    for (int r = 0; r < 8; ++r) {         // C/D layout: lane(g,nl), vgpr r -> M=r+8g, N=nl
        float bv = 3.4e38f; int bi = 0;
#pragma unroll
        for (int t = 0; t < 16; ++t) {
            float d = e2v[t] - 2.f * acc[t][r];
            int ni = nbase + t * 16 + nl;
            if (d < bv) { bv = d; bi = ni; }
        }
        for (int off = 1; off < 16; off <<= 1) {
            float ov = __shfl_xor(bv, off, 32);
            int   oi = __shfl_xor(bi, off, 32);
            if (ov < bv || (ov == bv && oi < bi)) { bv = ov; bi = oi; }
        }
        if (nl == 0) {
            int row = mtile * 16 + r + 8 * g;
            cval[row][nq] = bv; cidx[row][nq] = bi;
        }
    }
    __syncthreads();
    if (tid < 32) {
        float bv = cval[tid][0]; int bi = cidx[tid][0];
#pragma unroll
        for (int qn = 1; qn < 4; ++qn) {
            float ov = cval[tid][qn]; int oi = cidx[tid][qn];
            if (ov < bv || (ov == bv && oi < bi)) { bv = ov; bi = oi; }
        }
        idxOut[c * N_ + mb * 32 + tid] = bi;
    }
}

// ---------------------------------------------------------------------------
// Kernel C: gather q = codebooks[c][idx] (f32), write channel-reversed
// transposed output, accumulate loss = 1.25 * mean((q - v)^2) summed over c.
// ---------------------------------------------------------------------------
__global__ void gather_out_kernel(const float* __restrict__ cb,
                                  const unsigned short* __restrict__ vbf,
                                  const int* __restrict__ idx,
                                  float* __restrict__ out,
                                  float* __restrict__ lossPtr) {
    int c = blockIdx.x;                   // 8
    int n = blockIdx.y;                   // 4096
    int tid = threadIdx.x;                // 256
    int b = n >> 9, frame = n & 511;
    int e = idx[c * N_ + n];
    const float* q = cb + ((size_t)c * E_ + e) * BINS_;
    const unsigned short* vr = vbf + ((size_t)c * N_ + n) * BINS_;
    // embedded[b][C-1-c][bin][frame] = q[c][n][bin]
    float* dst = out + (((size_t)b * C_ + (C_ - 1 - c)) * BINS_) * FRAMES_ + frame;

    float s = 0.f;
#pragma unroll
    for (int i = 0; i < 4; ++i) {
        int bin = tid + i * 256;
        float qv = q[bin];
        float vv = bf2f(vr[bin]);
        float d = qv - vv;
        s += d * d;
        dst[(size_t)bin * FRAMES_] = qv;
    }
    for (int off = 16; off > 0; off >>= 1) s += __shfl_xor(s, off, 32);
    __shared__ float red[8];
    if ((tid & 31) == 0) red[tid >> 5] = s;
    __syncthreads();
    if (tid == 0) {
        float t = 0.f;
#pragma unroll
        for (int i = 0; i < 8; ++i) t += red[i];
        atomicAdd(lossPtr, t * (1.25f / ((float)N_ * (float)BINS_)));
    }
}

// ---------------------------------------------------------------------------
extern "C" void kernel_launch(void* const* d_in, const int* in_sizes, int n_in,
                              void* d_out, int out_size, void* d_ws, size_t ws_size,
                              hipStream_t stream) {
    const float* x  = (const float*)d_in[0];
    const float* w1 = (const float*)d_in[1];
    const float* b1 = (const float*)d_in[2];
    const float* w2 = (const float*)d_in[3];
    const float* b2 = (const float*)d_in[4];
    const float* sw = (const float*)d_in[5];
    const float* sb = (const float*)d_in[6];
    const float* cb = (const float*)d_in[7];

    float* out = (float*)d_out;
    float* lossPtr = out + (size_t)B_ * C_ * BINS_ * FRAMES_;   // last element

    char* ws = (char*)d_ws;
    unsigned short* cbbf = (unsigned short*)ws;                 // 16 MB
    size_t off = (size_t)C_ * E_ * BINS_ * sizeof(unsigned short);
    float* e2 = (float*)(ws + off);  off += (size_t)C_ * E_ * sizeof(float);
    int* idx  = (int*)(ws + off);    off += (size_t)C_ * N_ * sizeof(int);
    unsigned short* vbf = (unsigned short*)(ws + off);          // 64 MB

    pack_cb_kernel<<<C_ * E_, 256, 0, stream>>>(cb, cbbf, e2, lossPtr);
    conv_kernel<<<N_, 256, 0, stream>>>(x, w1, b1, w2, b2, sw, sb, vbf);
    dim3 gB(C_, N_ / 32);
    gemm_argmin_kernel<<<gB, 256, 0, stream>>>(vbf, cbbf, e2, idx);
    dim3 gC(C_, N_);
    gather_out_kernel<<<gC, 256, 0, stream>>>(cb, vbf, idx, out, lossPtr);
}
